// PCReaction_88441966559590
// MI455X (gfx1250) — compile-verified, matrix operand order checked by reading, not run on hardware
//
#include <hip/hip_runtime.h>

#define B_ 32
#define T_ 1024
#define D_ 256
#define V_ (T_*D_)
#define ETA_ 0.1f
#define INV_SQRT2_ 0.70710678118654752440f

typedef __bf16 bf16_t;
typedef bf16_t v8bf  __attribute__((ext_vector_type(8)));
typedef bf16_t v16bf __attribute__((ext_vector_type(16)));
typedef float  v8f   __attribute__((ext_vector_type(8)));

// Build a 16-element bf16 fragment from two contiguous 16-byte chunks.
__device__ __forceinline__ v16bf load_frag(const bf16_t* __restrict__ p0,
                                           const bf16_t* __restrict__ p1) {
    v8bf lo = *(const v8bf*)p0;
    v8bf hi = *(const v8bf*)p1;
    return __builtin_shufflevector(lo, hi, 0,1,2,3,4,5,6,7,8,9,10,11,12,13,14,15);
}

// ---------------- elementwise: coeffs = rho * node_w ----------------
__global__ void scale_kernel(const float* __restrict__ rho, const float* __restrict__ nw,
                             float* __restrict__ C) {
    int i = blockIdx.x * blockDim.x + threadIdx.x;       // exactly B_*V_ threads
    C[i] = rho[i] * nw[i & (V_ - 1)];
}

// ---------------- one inverse-Haar level along T --------------------
__global__ void idwt_level(const float* __restrict__ asrc, const float* __restrict__ C,
                           float* __restrict__ dst, int lgn) {
    int n = 1 << lgn;
    int i = blockIdx.x * blockDim.x + threadIdx.x;       // B_*n*D_ threads
    int d = i & (D_ - 1);
    int r = i >> 8;
    int j = r & (n - 1);
    int b = r >> lgn;
    float av = asrc[((size_t)(b * T_ + j)) * D_ + d];
    float dv = C[((size_t)(b * T_ + n + j)) * D_ + d];
    size_t o = ((size_t)(b * T_ + 2 * j)) * D_ + d;
    dst[o]      = (av + dv) * INV_SQRT2_;
    dst[o + D_] = (av - dv) * INV_SQRT2_;
}

// ---------------- one forward-Haar level along T --------------------
__global__ void dwt_level(const float* __restrict__ src, float* __restrict__ C,
                          float* __restrict__ adst, int lgn) {
    int n = 1 << lgn;
    int i = blockIdx.x * blockDim.x + threadIdx.x;       // B_*n*D_ threads
    int d = i & (D_ - 1);
    int r = i >> 8;
    int j = r & (n - 1);
    int b = r >> lgn;
    size_t si = ((size_t)(b * T_ + 2 * j)) * D_ + d;
    float e = src[si];
    float o = src[si + D_];
    C[((size_t)(b * T_ + n + j)) * D_ + d] = (e - o) * INV_SQRT2_;
    adst[((size_t)(b * T_ + j)) * D_ + d]  = (e + o) * INV_SQRT2_;
}

// ---------------- final: rho' = rho - ETA*(rho - C/node_w) ----------
__global__ void update_kernel(const float* __restrict__ rho_in, const float* __restrict__ C,
                              const float* __restrict__ nw, float* __restrict__ rho_out) {
    int i = blockIdx.x * blockDim.x + threadIdx.x;       // B_*V_ threads
    float pr = C[i] / nw[i & (V_ - 1)];
    float rv = rho_in[i];
    rho_out[i] = rv - ETA_ * (rv - pr);
}

// ---- one-time: W[band][k][n] -> transposed bf16 hi/lo Wt[band][n][k] ----
__global__ void convert_w(const float* __restrict__ W,
                          bf16_t* __restrict__ Whi, bf16_t* __restrict__ Wlo) {
    int i = blockIdx.x * blockDim.x + threadIdx.x;       // 11*256*256 threads
    int k = i & (D_ - 1);
    int r = i >> 8;
    int n = r & (D_ - 1);
    int band = r >> 8;
    float v = W[((size_t)band * D_ + k) * D_ + n];
    bf16_t h = (bf16_t)v;
    Whi[i] = h;
    Wlo[i] = (bf16_t)(v - (float)h);
}

// ---- per step: f32 signal -> bf16 hi/lo planes (row-major, same index) ----
__global__ void convert_sig(const float* __restrict__ sig,
                            bf16_t* __restrict__ Ahi, bf16_t* __restrict__ Alo) {
    int i = blockIdx.x * blockDim.x + threadIdx.x;       // B_*T_*D_ threads
    float v = sig[i];
    bf16_t h = (bf16_t)v;
    Ahi[i] = h;
    Alo[i] = (bf16_t)(v - (float)h);
}

// ---------------- head GEMM: rows t in [0,16) (mixed bands) ---------
__global__ void gemm_head(const float* __restrict__ sig, const float* __restrict__ W,
                          const float* __restrict__ bias, float* __restrict__ out) {
    int i = blockIdx.x * blockDim.x + threadIdx.x;       // B_*16*D_ threads
    int e = i & (D_ - 1);
    int r = i >> 8;
    int t = r & 15;
    int b = r >> 4;
    int band = (t == 0) ? 0 : (32 - __builtin_clz((unsigned)t));
    const float* Wp  = W + (size_t)band * D_ * D_;
    const float* row = sig + ((size_t)(b * T_ + t)) * D_;
    float acc = bias[band * D_ + e];
    for (int k = 0; k < D_; ++k) acc += row[k] * Wp[k * D_ + e];
    out[((size_t)(b * T_ + t)) * D_ + e] = acc;
}

// ---------------- main GEMM: bf16 hi/lo split WMMA, t >= 16 ---------
// One wave computes a 16x64 output strip: 1 M-tile x 4 N-tiles, K=256.
// A fragment (16-bit 16x32 layout): lane&15 = M row, half-wave = K sub-block:
//   elem j -> K = k0 + (j>>3)*16 + (lane>>4)*8 + (j&7)
// B stored pre-transposed (Wt[band][n][k]) so a lane's column fragment is two
// contiguous b128 loads with the identical K pattern. 4 independent
// accumulators -> no WMMA->WMMA accumulation RAW between consecutive ops.
__global__ void __launch_bounds__(128) gemm_wmma(const bf16_t* __restrict__ Ahi,
                                                 const bf16_t* __restrict__ Alo,
                                                 const bf16_t* __restrict__ Whi,
                                                 const bf16_t* __restrict__ Wlo,
                                                 const float* __restrict__ bias,
                                                 float* __restrict__ out) {
    int wave  = blockIdx.x * (blockDim.x >> 5) + (threadIdx.x >> 5);
    int lane  = threadIdx.x & 31;
    int l15   = lane & 15;
    int hh    = lane >> 4;              // half-wave -> K sub-block
    int ng    = wave & 3;               // N group: columns [ng*64, ng*64+64)
    int mtile = wave >> 2;              // 0 .. 32*63-1
    int b     = mtile / 63;
    int ti    = mtile - b * 63;
    int t0    = 16 + ti * 16;           // 16-aligned, single band per tile
    int band  = 32 - __builtin_clz((unsigned)t0);
    int m0    = b * T_ + t0;

    const bf16_t* ArH = Ahi + ((size_t)(m0 + l15)) * D_;
    const bf16_t* ArL = Alo + ((size_t)(m0 + l15)) * D_;

    const bf16_t* BrH[4];
    const bf16_t* BrL[4];
    v8f acc[4];
    #pragma unroll
    for (int nt = 0; nt < 4; ++nt) {
        int nc = ng * 64 + nt * 16 + l15;
        size_t wo = ((size_t)(band * D_ + nc)) * D_;
        BrH[nt] = Whi + wo;
        BrL[nt] = Wlo + wo;
        float bv = bias[band * D_ + nc];
        acc[nt] = (v8f){bv, bv, bv, bv, bv, bv, bv, bv};
    }

    for (int k0 = 0; k0 < D_; k0 += 32) {
        int o0 = k0 + hh * 8;
        int o1 = k0 + 16 + hh * 8;
        v16bf ahi = load_frag(ArH + o0, ArH + o1);
        v16bf alo = load_frag(ArL + o0, ArL + o1);
        #pragma unroll
        for (int nt = 0; nt < 4; ++nt) {
            v16bf bhi = load_frag(BrH[nt] + o0, BrH[nt] + o1);
            v16bf blo = load_frag(BrL[nt] + o0, BrL[nt] + o1);
            acc[nt] = __builtin_amdgcn_wmma_f32_16x16x32_bf16(false, ahi, false, bhi,
                                                              (short)0, acc[nt], false, false);
            acc[nt] = __builtin_amdgcn_wmma_f32_16x16x32_bf16(false, ahi, false, blo,
                                                              (short)0, acc[nt], false, false);
            acc[nt] = __builtin_amdgcn_wmma_f32_16x16x32_bf16(false, alo, false, bhi,
                                                              (short)0, acc[nt], false, false);
        }
    }

    // C/D layout: elem r -> row M = r + 8*hh, col N = lane&15
    #pragma unroll
    for (int nt = 0; nt < 4; ++nt) {
        int nc = ng * 64 + nt * 16 + l15;
        #pragma unroll
        for (int r = 0; r < 8; ++r) {
            out[((size_t)(m0 + r + hh * 8)) * D_ + nc] = acc[nt][r];
        }
    }
}

extern "C" void kernel_launch(void* const* d_in, const int* in_sizes, int n_in,
                              void* d_out, int out_size, void* d_ws, size_t ws_size,
                              hipStream_t stream) {
    (void)in_sizes; (void)n_in; (void)out_size; (void)ws_size;
    const float* rho0 = (const float*)d_in[0];
    const float* nw   = (const float*)d_in[1];
    const float* W    = (const float*)d_in[2];
    const float* bias = (const float*)d_in[3];
    float* rho_out    = (float*)d_out;

    char* ws = (char*)d_ws;
    const size_t BUF  = (size_t)B_ * T_ * D_ * sizeof(float);     // 32 MB
    const size_t WELT = (size_t)11 * D_ * D_;                     // 720896
    float* C  = (float*)(ws);
    float* P0 = (float*)(ws + BUF);
    float* P1 = (float*)(ws + 2 * BUF);
    bf16_t* Whi = (bf16_t*)(ws + 3 * BUF);
    bf16_t* Wlo = Whi + WELT;
    // bf16 signal planes overlay C (dead between IDWT-end and DWT-start)
    bf16_t* Ahi = (bf16_t*)C;
    bf16_t* Alo = Ahi + (size_t)B_ * T_ * D_;

    // one-time: transpose+split W to bf16 hi/lo
    convert_w<<<(int)(WELT / 256), 256, 0, stream>>>(W, Whi, Wlo);

    const int GEMM_BLOCKS = (32 * 63 * 4) / 4;                    // 8064 waves, 4/block

    for (int step = 0; step < 2; ++step) {
        const float* rho_src = (step == 0) ? rho0 : (const float*)rho_out;

        // coeffs = rho * node_w
        scale_kernel<<<(B_ * V_) / 256, 256, 0, stream>>>(rho_src, nw, C);

        // IDWT: 10 levels, approx ping-pongs P0/P1, details read from C
        float* ping[2] = {P0, P1};
        for (int i = 0; i < 10; ++i) {
            int n = 1 << i;
            const float* asrc = (i == 0) ? (const float*)C : (const float*)ping[(i + 1) & 1];
            idwt_level<<<(B_ * n * D_) / 256, 256, 0, stream>>>(asrc, C, ping[i & 1], i);
        }
        // signal now in P1; C is dead -> reuse as bf16 planes
        convert_sig<<<(B_ * T_ * D_) / 256, 256, 0, stream>>>(P1, Ahi, Alo);

        // per-band Linear(D->D): WMMA body (t>=16) + scalar head (t<16)
        gemm_wmma<<<GEMM_BLOCKS, 128, 0, stream>>>(Ahi, Alo, Whi, Wlo, bias, P0);
        gemm_head<<<(B_ * 16 * D_) / 256, 256, 0, stream>>>(P1, W, bias, P0);
        // predicted signal now in P0

        // DWT: 10 levels, approx ping-pongs P1/P0; last level drops approx in C row 0
        float* pong[2] = {P1, P0};
        for (int i = 0; i < 10; ++i) {
            int lgn = 9 - i;
            int n = 1 << lgn;
            const float* src = (i == 0) ? (const float*)P0 : (const float*)pong[(i + 1) & 1];
            float* adst = (i == 9) ? C : pong[i & 1];
            dwt_level<<<(B_ * n * D_) / 256, 256, 0, stream>>>(src, C, adst, lgn);
        }

        // rho' = rho - ETA*(rho - C/node_w)
        update_kernel<<<(B_ * V_) / 256, 256, 0, stream>>>(rho_src, C, nw, rho_out);
    }
}
